// SSRAttnProcessor_42743514530014
// MI455X (gfx1250) — matrix operand order repaired and checked by reading, not compiled, not run on hardware
//
#include <hip/hip_runtime.h>
#include <hip/hip_bf16.h>

typedef __bf16 bf16_t;
typedef __attribute__((ext_vector_type(4)))  __bf16 v4bf;
typedef __attribute__((ext_vector_type(8)))  __bf16 v8bf;
typedef __attribute__((ext_vector_type(16))) __bf16 v16bf;
typedef __attribute__((ext_vector_type(8)))  float  v8f;
typedef __attribute__((ext_vector_type(4)))  unsigned int v4u;
typedef __attribute__((ext_vector_type(8)))  unsigned int v8u;

static __device__ __forceinline__ v8f zero_v8f() {
    v8f z = {0.f,0.f,0.f,0.f,0.f,0.f,0.f,0.f};
    return z;
}
static __device__ __forceinline__ v8bf zero_v8bf() {
    v8bf z = {(__bf16)0.f,(__bf16)0.f,(__bf16)0.f,(__bf16)0.f,
              (__bf16)0.f,(__bf16)0.f,(__bf16)0.f,(__bf16)0.f};
    return z;
}

// A-operand fragment: lane holds K = {base..base+7} U {base+16..base+23}
static __device__ __forceinline__ v16bf fragA_ld(const bf16_t* p) {
    v8bf lo = *(const v8bf*)p;
    v8bf hi = *(const v8bf*)(p + 16);
    return __builtin_shufflevector(lo, hi, 0,1,2,3,4,5,6,7,8,9,10,11,12,13,14,15);
}
// B-operand fragment: lane holds K = {base..base+7} U {base+16..base+23}
static __device__ __forceinline__ v16bf fragB_ld(const bf16_t* p) {
    v8bf lo = *(const v8bf*)p;
    v8bf hi = *(const v8bf*)(p + 16);
    return __builtin_shufflevector(lo, hi, 0,1,2,3,4,5,6,7,8,9,10,11,12,13,14,15);
}

// LDS byte offset of a generic pointer (addrspacecast to LDS, truncate)
static __device__ __forceinline__ unsigned lds_off_of(const void* p) {
    return (unsigned)(unsigned long long)(const __attribute__((address_space(3))) void*)p;
}

// ---------------------------------------------------------------------------
// Tensor Data Mover: 2D tile of bf16 (2-byte) elements -> LDS, with LDS
// padding of 16B after every 64B so rows land at stride 40 bf16 (80B).
// D# per CDNA5 ISA. 2 SGPR groups (2D tensor).
// Issued per-wave (EXEC ignored); caller gates to one wave per block.
// ---------------------------------------------------------------------------
static __device__ __forceinline__ void tdm_load_tile_bf16(
    unsigned lds_addr,                 // LDS byte address of tile start
    const bf16_t* gsrc,                // global address of tile start
    unsigned rem_d0, unsigned rem_d1,  // remaining tensor extent (elements)
    unsigned stride_d0)                // line stride (elements)
{
    unsigned long long ga = (unsigned long long)gsrc;
    v4u g0;
    g0[0] = 1u;                                            // count=1, user mode
    g0[1] = lds_addr;                                      // lds_addr
    g0[2] = (unsigned)ga;                                  // global_addr[31:0]
    g0[3] = (unsigned)((ga >> 32) & 0x01FFFFFFu)           // global_addr[56:32]
          | (2u << 30);                                    // type=2 ("image")
    v8u g1;
    g1[0] = (1u << 16)      // data_size = 1 -> 2 bytes
          | (1u << 20)      // pad_enable
          | (3u << 22)      // pad_interval code 3 -> every 16 DWORDs (64B)
          | (3u << 25);     // pad_amount  code 3 -> 4 DWORDs (16B)
    g1[1] = (rem_d0 & 0xFFFFu) << 16;                      // tensor_dim0[15:0]
    g1[2] = (rem_d0 >> 16) | ((rem_d1 & 0xFFFFu) << 16);   // dim0 hi | dim1 lo
    g1[3] = (rem_d1 >> 16) | (32u << 16);                  // dim1 hi | tile_dim0=32
    g1[4] = 128u;                                          // tile_dim1=128, tile_dim2=0
    g1[5] = stride_d0;                                     // tensor_dim0_stride lo
    g1[6] = 0u;                                            // stride hi | dim1_stride lo
    g1[7] = 0u;                                            // dim1_stride hi
    asm volatile("tensor_load_to_lds %0, %1" :: "s"(g0), "s"(g1) : "memory");
}

static __device__ __forceinline__ void wait_tensorcnt0() {
    asm volatile("s_wait_tensorcnt 0x0" ::: "memory");
}

// ---------------------------------------------------------------------------
// elementwise f32 -> bf16 conversion (weights)
// ---------------------------------------------------------------------------
__global__ void cvt_f32_bf16(const float* __restrict__ in, bf16_t* __restrict__ out, int n) {
    for (int i = blockIdx.x * blockDim.x + threadIdx.x; i < n; i += gridDim.x * blockDim.x)
        out[i] = (bf16_t)in[i];
}

// ---------------------------------------------------------------------------
// Repack K into [B][80][1280] and V transposed into [B][1280][96] (pads zeroed)
// ---------------------------------------------------------------------------
__global__ void repack_kv(const bf16_t* __restrict__ Kbf, const bf16_t* __restrict__ Vbf,
                          bf16_t* __restrict__ Kp, bf16_t* __restrict__ Vt) {
    const int nK = 16 * 80 * 1280;
    const int nV = 16 * 1280 * 96;
    for (int i = blockIdx.x * blockDim.x + threadIdx.x; i < nK + nV;
         i += gridDim.x * blockDim.x) {
        if (i < nK) {
            int b = i / (80 * 1280); int r = i % (80 * 1280);
            int t = r / 1280;        int c = r % 1280;
            Kp[i] = (t < 77) ? Kbf[((size_t)b * 77 + t) * 1280 + c] : (bf16_t)0.f;
        } else {
            int j = i - nK;
            int b = j / (1280 * 96); int r = j % (1280 * 96);
            int c = r / 96;          int t = r % 96;
            Vt[j] = (t < 77) ? Vbf[((size_t)b * 77 + t) * 1280 + c] : (bf16_t)0.f;
        }
    }
}

// ---------------------------------------------------------------------------
// Tiled WMMA GEMM: C[M,N] = A[M,K] * B[N,K]^T  (torch Linear convention)
//   A: f32 (manual convert to bf16 in LDS) or bf16 (staged via TDM);
//   B: bf16 [N,K] row-major, staged via TDM.  Double-buffered LDS ping-pong.
// Block tile 128x128x32, 256 threads = 8 waves, each wave 32x64.
// ---------------------------------------------------------------------------
template<bool A_F32, bool OUT_F32>
__global__ void __launch_bounds__(256) gemm_bf16_wmma(
    const void* __restrict__ Aptr, int lda,
    const bf16_t* __restrict__ B, int ldb,
    void* __restrict__ Cptr, int ldc,
    const float* __restrict__ bias,
    int M, int N, int K)
{
    __shared__ bf16_t As[2][128 * 40];   // stride 40 bf16 = 80B rows (16B aligned)
    __shared__ bf16_t Bs[2][128 * 40];

    const int tid  = threadIdx.x;
    const int lane = tid & 31, wave = tid >> 5;
    const int g = lane >> 4, ln = lane & 15;
    const int wm = wave & 3, wn = wave >> 2;
    const int row0 = blockIdx.y * 128;
    const int col0 = blockIdx.x * 128;

    v8f acc[2][4];
    #pragma unroll
    for (int i = 0; i < 2; ++i)
        #pragma unroll
        for (int j = 0; j < 4; ++j) acc[i][j] = zero_v8f();

    auto stage = [&](int kt, int bufi) {
        const int k0 = kt << 5;
        if constexpr (A_F32) {
            // manual f32 -> bf16 staging (handles M edge)
            const float* A = (const float*)Aptr;
            #pragma unroll
            for (int i = 0; i < 4; ++i) {
                int seg = tid + i * 256;                // 1024 float4 segments
                int r = seg >> 3, c4 = (seg & 7) << 2;
                float4 av = make_float4(0.f, 0.f, 0.f, 0.f);
                int gr = row0 + r;
                if (gr < M) av = *(const float4*)(A + (size_t)gr * lda + k0 + c4);
                v4bf bv = {(bf16_t)av.x, (bf16_t)av.y, (bf16_t)av.z, (bf16_t)av.w};
                *(v4bf*)(&As[bufi][0] + r * 40 + c4) = bv;
            }
            if (wave == 0) {
                tdm_load_tile_bf16(lds_off_of(&Bs[bufi][0]),
                                   B + (size_t)col0 * ldb + k0,
                                   (unsigned)(ldb - k0), (unsigned)(N - col0),
                                   (unsigned)ldb);
            }
        } else {
            if (wave == 0) {
                const bf16_t* A = (const bf16_t*)Aptr;
                tdm_load_tile_bf16(lds_off_of(&As[bufi][0]),
                                   A + (size_t)row0 * lda + k0,
                                   (unsigned)(lda - k0), (unsigned)(M - row0),
                                   (unsigned)lda);
                tdm_load_tile_bf16(lds_off_of(&Bs[bufi][0]),
                                   B + (size_t)col0 * ldb + k0,
                                   (unsigned)(ldb - k0), (unsigned)(N - col0),
                                   (unsigned)ldb);
            }
        }
    };

    const int nk = K >> 5;
    stage(0, 0);
    for (int kt = 0; kt < nk; ++kt) {
        const int buf = kt & 1;
        wait_tensorcnt0();          // no-op for waves with TENSORcnt==0
        __syncthreads();            // stage(kt) visible to all waves
        if (kt + 1 < nk) stage(kt + 1, buf ^ 1);   // overlap DMA with math

        v16bf afrag[2];
        #pragma unroll
        for (int tm = 0; tm < 2; ++tm)
            afrag[tm] = fragA_ld(&As[buf][0] + (wm * 32 + tm * 16 + ln) * 40 + g * 8);
        #pragma unroll
        for (int tn = 0; tn < 4; ++tn) {
            v16bf bfrag = fragB_ld(&Bs[buf][0] + (wn * 64 + tn * 16 + ln) * 40 + g * 8);
            #pragma unroll
            for (int tm = 0; tm < 2; ++tm)
                acc[tm][tn] = __builtin_amdgcn_wmma_f32_16x16x32_bf16(
                    false, afrag[tm], false, bfrag, (short)0, acc[tm][tn], false, false);
        }
    }

    // ---- epilogue ----
    #pragma unroll
    for (int tm = 0; tm < 2; ++tm) {
        #pragma unroll
        for (int v = 0; v < 8; ++v) {
            int grow = row0 + wm * 32 + tm * 16 + v + 8 * g;
            if (grow >= M) continue;
            #pragma unroll
            for (int tn = 0; tn < 4; ++tn) {
                int gcol = col0 + wn * 64 + tn * 16 + ln;
                float val = acc[tm][tn][v];
                if constexpr (OUT_F32) {
                    if (bias) val += bias[gcol];
                    ((float*)Cptr)[(size_t)grow * ldc + gcol] = val;
                } else {
                    ((bf16_t*)Cptr)[(size_t)grow * ldc + gcol] = (bf16_t)val;
                }
            }
        }
    }
}

// ---------------------------------------------------------------------------
// Fused attention: per (s-block of 128, head, batch). Q read / out written
// in place on the same [rows, head-slice] region of QO (disjoint per block).
// ---------------------------------------------------------------------------
__global__ void __launch_bounds__(256) attn_wmma(
    bf16_t* __restrict__ QO,        // [B*4096, 1280] : Q in, attn-out out
    const bf16_t* __restrict__ Kp,  // [B, 80, 1280]
    const bf16_t* __restrict__ Vt)  // [B, 1280, 96]
{
    __shared__ bf16_t Psh[8 * 16 * 104];   // per-wave 16x96 P (stride 104)

    const int tid  = threadIdx.x;
    const int lane = tid & 31, wave = tid >> 5;
    const int g = lane >> 4, ln = lane & 15;
    const int sb = blockIdx.x, h = blockIdx.y, b = blockIdx.z;
    const size_t qbase = ((size_t)b * 4096 + sb * 128 + wave * 16) * 1280;
    const bf16_t* Kb = Kp + (size_t)b * 80 * 1280;
    const bf16_t* Vb = Vt + (size_t)b * 1280 * 96;
    const int hc = h * 160;

    // ---- scores: 16x80 tile = Qh(16x160) @ Kh^T ----
    v8f sc[5];
    #pragma unroll
    for (int j = 0; j < 5; ++j) sc[j] = zero_v8f();
    #pragma unroll
    for (int kt = 0; kt < 5; ++kt) {
        v16bf a = fragA_ld(QO + qbase + (size_t)ln * 1280 + hc + kt * 32 + g * 8);
        #pragma unroll
        for (int nt = 0; nt < 5; ++nt) {
            v16bf bf = fragB_ld(Kb + (size_t)(nt * 16 + ln) * 1280 + hc + kt * 32 + g * 8);
            sc[nt] = __builtin_amdgcn_wmma_f32_16x16x32_bf16(
                false, a, false, bf, (short)0, sc[nt], false, false);
        }
    }

    // ---- softmax over t (mask t>=77), write P (bf16) to LDS ----
    const float scale = 0.07905694150420949f;   // 160^-0.5
    bf16_t* Pw = Psh + wave * 16 * 104;
    #pragma unroll
    for (int v = 0; v < 8; ++v) {
        float m = -3.4e38f;
        #pragma unroll
        for (int nt = 0; nt < 5; ++nt) {
            float s = sc[nt][v] * scale;
            if (nt * 16 + ln >= 77) s = -3.4e38f;
            sc[nt][v] = s;
            m = fmaxf(m, s);
        }
        m = fmaxf(m, __shfl_xor(m, 8, 32));
        m = fmaxf(m, __shfl_xor(m, 4, 32));
        m = fmaxf(m, __shfl_xor(m, 2, 32));
        m = fmaxf(m, __shfl_xor(m, 1, 32));
        float s = 0.f;
        #pragma unroll
        for (int nt = 0; nt < 5; ++nt) {
            float p = __expf(sc[nt][v] - m);
            sc[nt][v] = p;
            s += p;
        }
        s += __shfl_xor(s, 8, 32);
        s += __shfl_xor(s, 4, 32);
        s += __shfl_xor(s, 2, 32);
        s += __shfl_xor(s, 1, 32);
        float rs = 1.0f / s;
        #pragma unroll
        for (int nt = 0; nt < 5; ++nt)
            Pw[(v + 8 * g) * 104 + nt * 16 + ln] = (bf16_t)(sc[nt][v] * rs);
        Pw[(v + 8 * g) * 104 + 80 + ln] = (bf16_t)0.f;   // zero pad t=80..95
    }
    __syncthreads();

    // ---- out: 16x160 = P(16x96) @ Vh(96x160), pad rows of V are zero ----
    v8f oa[10];
    #pragma unroll
    for (int j = 0; j < 10; ++j) oa[j] = zero_v8f();
    #pragma unroll
    for (int kt = 0; kt < 3; ++kt) {
        v16bf a = fragA_ld(Pw + ln * 104 + kt * 32 + g * 8);
        #pragma unroll
        for (int nt = 0; nt < 10; ++nt) {
            v16bf bf = fragB_ld(Vb + (size_t)(hc + nt * 16 + ln) * 96 + kt * 32 + g * 8);
            oa[nt] = __builtin_amdgcn_wmma_f32_16x16x32_bf16(
                false, a, false, bf, (short)0, oa[nt], false, false);
        }
    }
    #pragma unroll
    for (int nt = 0; nt < 10; ++nt)
        #pragma unroll
        for (int v = 0; v < 8; ++v)
            QO[qbase + (size_t)(v + 8 * g) * 1280 + hc + nt * 16 + ln] = (bf16_t)oa[nt][v];
}

// ---------------------------------------------------------------------------
extern "C" void kernel_launch(void* const* d_in, const int* in_sizes, int n_in,
                              void* d_out, int out_size, void* d_ws, size_t ws_size,
                              hipStream_t stream)
{
    (void)in_sizes; (void)n_in; (void)out_size; (void)ws_size;
    const float* hs  = (const float*)d_in[0];   // [16,4096,1280]
    const float* ehs = (const float*)d_in[1];   // [16,77,768]
    const float* wq  = (const float*)d_in[2];   // [1280,1280]
    const float* wk  = (const float*)d_in[3];   // [1280,768]
    const float* wv  = (const float*)d_in[4];   // [1280,768]
    const float* wo  = (const float*)d_in[5];   // [1280,1280]
    const float* bo  = (const float*)d_in[6];   // [1280]

    char* ws = (char*)d_ws;
    size_t off = 0;
    auto alloc = [&](size_t bytes) -> void* {
        void* p = ws + off;
        off = (off + bytes + 255) & ~(size_t)255;
        return p;
    };
    bf16_t* Wq  = (bf16_t*)alloc((size_t)1280 * 1280 * 2);
    bf16_t* Wk  = (bf16_t*)alloc((size_t)1280 * 768  * 2);
    bf16_t* Wv  = (bf16_t*)alloc((size_t)1280 * 768  * 2);
    bf16_t* Wo  = (bf16_t*)alloc((size_t)1280 * 1280 * 2);
    bf16_t* Kbf = (bf16_t*)alloc((size_t)1232 * 1280 * 2);
    bf16_t* Vbf = (bf16_t*)alloc((size_t)1232 * 1280 * 2);
    bf16_t* Kp  = (bf16_t*)alloc((size_t)16 * 80 * 1280 * 2);
    bf16_t* Vt  = (bf16_t*)alloc((size_t)16 * 1280 * 96 * 2);
    bf16_t* QO  = (bf16_t*)alloc((size_t)65536 * 1280 * 2);

    // 1) weights -> bf16
    cvt_f32_bf16<<<1024, 256, 0, stream>>>(wq, Wq, 1280 * 1280);
    cvt_f32_bf16<<<1024, 256, 0, stream>>>(wk, Wk, 1280 * 768);
    cvt_f32_bf16<<<1024, 256, 0, stream>>>(wv, Wv, 1280 * 768);
    cvt_f32_bf16<<<1024, 256, 0, stream>>>(wo, Wo, 1280 * 1280);

    // 2) K,V projections: [1232,768] @ [1280,768]^T -> bf16 [1232,1280]
    gemm_bf16_wmma<true,  false><<<dim3(10, 10),  256, 0, stream>>>(
        ehs, 768, Wk, 768, Kbf, 1280, nullptr, 1232, 1280, 768);
    gemm_bf16_wmma<true,  false><<<dim3(10, 10),  256, 0, stream>>>(
        ehs, 768, Wv, 768, Vbf, 1280, nullptr, 1232, 1280, 768);

    // 3) repack K (t-pad to 80) and transpose V (t-pad to 96, zeroed)
    repack_kv<<<2048, 256, 0, stream>>>(Kbf, Vbf, Kp, Vt);

    // 4) Q projection: [65536,1280] @ [1280,1280]^T -> bf16 QO
    gemm_bf16_wmma<true,  false><<<dim3(10, 512), 256, 0, stream>>>(
        hs, 1280, Wq, 1280, QO, 1280, nullptr, 65536, 1280, 1280);

    // 5) fused attention (in place on QO): grid = (S/128, H, B)
    attn_wmma<<<dim3(32, 8, 16), 256, 0, stream>>>(QO, Kp, Vt);

    // 6) output projection + bias -> f32 d_out
    gemm_bf16_wmma<false, true ><<<dim3(10, 512), 256, 0, stream>>>(
        QO, 1280, Wo, 1280, d_out, 1280, bo, 65536, 1280, 1280);
}
